// FlaxPhiMoeDecoderLayer_17446157156567
// MI455X (gfx1250) — compile-verified, matrix operand order checked by reading, not run on hardware
//
#include <hip/hip_runtime.h>

// ---------------------------------------------------------------------------
// Problem constants (FlaxPhiMoeDecoderLayer)
// ---------------------------------------------------------------------------
constexpr int Bsz = 2, Ssz = 1024, Hsz = 2048, NHn = 16, NKVn = 4, HDn = 128;
constexpr int Fsz = 4096, En = 8, TOPKn = 2;
constexpr int Tn  = Bsz * Ssz;      // 2048 tokens
constexpr int QN  = NHn * HDn;      // 2048
constexpr int KVN = NKVn * HDn;     // 512
constexpr int NAn = Tn * TOPKn;     // 4096 expert assignments
constexpr float EPSf = 1e-5f;
constexpr float ATT_SCALE = 0.08838834764831845f; // 1/sqrt(128)

typedef __attribute__((ext_vector_type(16))) __bf16 v16bf;
typedef __attribute__((ext_vector_type(8)))  __bf16 v8bf;
typedef __attribute__((ext_vector_type(8)))  float  v8f;

#define DEVI __device__ __forceinline__

DEVI v8f zero8() {
  v8f z;
#pragma unroll
  for (int i = 0; i < 8; ++i) z[i] = 0.f;
  return z;
}

// Load one 16x32 bf16 A/B fragment for v_wmma_f32_16x16x32_bf16.
// Lane's 16 elements: K = half*8 + [0..7] and K = 16 + half*8 + [0..7].
DEVI v16bf ld_frag(const __bf16* rowptr, int k, int half) {
  const __bf16* p = rowptr + k + half * 8;
  v8bf lo = *(const v8bf*)(p);
  v8bf hi = *(const v8bf*)(p + 16);
  v16bf f;
#pragma unroll
  for (int i = 0; i < 8; ++i) { f[i] = lo[i]; f[8 + i] = hi[i]; }
  return f;
}

DEVI v8f wmma_bf16(v16bf a, v16bf b, v8f c) {
  return __builtin_amdgcn_wmma_f32_16x16x32_bf16(false, a, false, b, (short)0, c,
                                                 false, false);
}

// CDNA5 async global->LDS copy of one 16B chunk per lane (ASYNCcnt tracked).
DEVI void async_cp16(const __bf16* gsrc, __bf16* ldst) {
  unsigned lds = (unsigned)(uintptr_t)ldst;
  asm volatile("global_load_async_to_lds_b128 %0, %1, off"
               :: "v"(lds), "v"(gsrc)
               : "memory");
}
DEVI void wait_async0() { asm volatile("s_wait_asynccnt 0" ::: "memory"); }

// Plain staging copy (global_load_b128 + ds_store_b128).
DEVI void cp16(const __bf16* gsrc, __bf16* ldst) {
  *(v8bf*)ldst = *(const v8bf*)gsrc;
}

// ---------------------------------------------------------------------------
// Weight transpose + fp32 -> bf16:  in [K][N] (row major) -> out [N][K]
// grid: (N/32, K/32, batch)  block: (32, 8)
// ---------------------------------------------------------------------------
__global__ __launch_bounds__(256) void k_transpose_cvt(const float* __restrict__ in,
                                                       __bf16* __restrict__ out,
                                                       int K, int N) {
  __shared__ float tile[32][33];
  const int nb = blockIdx.x * 32, kb = blockIdx.y * 32;
  const float* src = in + (size_t)blockIdx.z * K * N;
  __bf16* dst = out + (size_t)blockIdx.z * K * N;
#pragma unroll
  for (int i = threadIdx.y; i < 32; i += 8)
    tile[i][threadIdx.x] = src[(size_t)(kb + i) * N + nb + threadIdx.x];
  __syncthreads();
#pragma unroll
  for (int i = threadIdx.y; i < 32; i += 8)
    dst[(size_t)(nb + i) * K + kb + threadIdx.x] = (__bf16)tile[threadIdx.x][i];
}

// ---------------------------------------------------------------------------
// LayerNorm (H=2048) fp32 in -> bf16 out.  grid: T blocks of 256.
// ---------------------------------------------------------------------------
__global__ __launch_bounds__(256) void k_layernorm_bf16(const float* __restrict__ x,
                                                        const float* __restrict__ w,
                                                        const float* __restrict__ b,
                                                        __bf16* __restrict__ y) {
  const int t = blockIdx.x, tid = threadIdx.x;
  const float* xp = x + (size_t)t * Hsz;
  float vals[8], s = 0.f, s2 = 0.f;
#pragma unroll
  for (int i = 0; i < 8; ++i) {
    float v = xp[tid + i * 256];
    vals[i] = v; s += v; s2 += v * v;
  }
#pragma unroll
  for (int off = 16; off; off >>= 1) {
    s += __shfl_xor(s, off, 32);
    s2 += __shfl_xor(s2, off, 32);
  }
  __shared__ float rs[8], rs2[8];
  const int wid = tid >> 5, lane = tid & 31;
  if (lane == 0) { rs[wid] = s; rs2[wid] = s2; }
  __syncthreads();
  float ts = 0.f, ts2 = 0.f;
#pragma unroll
  for (int i = 0; i < 8; ++i) { ts += rs[i]; ts2 += rs2[i]; }
  const float mu = ts / Hsz;
  const float var = ts2 / Hsz - mu * mu;
  const float inv = rsqrtf(var + EPSf);
#pragma unroll
  for (int i = 0; i < 8; ++i) {
    int c = tid + i * 256;
    y[(size_t)t * Hsz + c] = (__bf16)((vals[i] - mu) * inv * w[c] + b[c]);
  }
}

// ---------------------------------------------------------------------------
// Generic bf16 WMMA GEMM:  C[M,N] = A[M,K] * Bt[N,K]^T (+bias)
// 128x128 block tile, 8 waves (wave = 32x64 = 2x4 WMMA sub-tiles).
// A/B tiles double-buffered in LDS, filled with async global->LDS copies so
// the k+1 fill overlaps the k WMMAs; s_wait_asynccnt 0 + barrier per step.
// ---------------------------------------------------------------------------
template <bool BIAS>
__global__ __launch_bounds__(256) void k_gemm_bf16_f32(const __bf16* __restrict__ A,
                                                       const __bf16* __restrict__ Bt,
                                                       const float* __restrict__ bias,
                                                       float* __restrict__ C,
                                                       int N, int Kd) {
  __shared__ __align__(16) __bf16 sA[2][128 * 32];
  __shared__ __align__(16) __bf16 sB[2][128 * 32];
  const int tid = threadIdx.x, wid = tid >> 5, lane = tid & 31;
  const int half = lane >> 4, ln = lane & 15;
  const int wm = wid & 3, wn = wid >> 2;
  const int mBlk = blockIdx.y * 128, nBlk = blockIdx.x * 128;

  // staging indices: 256 threads x 16B cover a 64x32 half-tile per shot
  const int crow = tid >> 2;          // 0..63
  const int cch  = (tid & 3) * 8;     // element offset of 16B chunk
  const __bf16* gA0 = A + (size_t)(mBlk + crow) * Kd + cch;
  const __bf16* gA1 = A + (size_t)(mBlk + 64 + crow) * Kd + cch;
  const __bf16* gB0 = Bt + (size_t)(nBlk + crow) * Kd + cch;
  const __bf16* gB1 = Bt + (size_t)(nBlk + 64 + crow) * Kd + cch;

  v8f acc[2][4];
#pragma unroll
  for (int i = 0; i < 2; ++i)
#pragma unroll
    for (int j = 0; j < 4; ++j) acc[i][j] = zero8();

  // prologue fill of buffer 0
  async_cp16(gA0, &sA[0][crow * 32 + cch]);
  async_cp16(gA1, &sA[0][(64 + crow) * 32 + cch]);
  async_cp16(gB0, &sB[0][crow * 32 + cch]);
  async_cp16(gB1, &sB[0][(64 + crow) * 32 + cch]);
  wait_async0();
  __syncthreads();

  int buf = 0;
  for (int k = 0; k < Kd; k += 32) {
    if (k + 32 < Kd) {
      const int nb = buf ^ 1, kn = k + 32;
      async_cp16(gA0 + kn, &sA[nb][crow * 32 + cch]);
      async_cp16(gA1 + kn, &sA[nb][(64 + crow) * 32 + cch]);
      async_cp16(gB0 + kn, &sB[nb][crow * 32 + cch]);
      async_cp16(gB1 + kn, &sB[nb][(64 + crow) * 32 + cch]);
    }
    const __bf16* pa0 = &sA[buf][(wm * 32 + ln) * 32];
    v16bf a0 = ld_frag(pa0, 0, half);
    v16bf a1 = ld_frag(pa0 + 16 * 32, 0, half);
#pragma unroll
    for (int j = 0; j < 4; ++j) {
      v16bf bfr = ld_frag(&sB[buf][(wn * 64 + j * 16 + ln) * 32], 0, half);
      acc[0][j] = wmma_bf16(a0, bfr, acc[0][j]);
      acc[1][j] = wmma_bf16(a1, bfr, acc[1][j]);
    }
    wait_async0();      // async fills done before anyone touches next buffer
    __syncthreads();    // and all waves done reading current buffer
    buf ^= 1;
  }
#pragma unroll
  for (int i = 0; i < 2; ++i)
#pragma unroll
    for (int j = 0; j < 4; ++j)
#pragma unroll
      for (int r = 0; r < 8; ++r) {
        int row = mBlk + wm * 32 + i * 16 + half * 8 + r;
        int col = nBlk + wn * 64 + j * 16 + ln;
        float v = acc[i][j][r];
        if (BIAS) v += bias[col];
        C[(size_t)row * N + col] = v;
      }
}

// ---------------------------------------------------------------------------
// RoPE + repack to bf16 head-major layouts.
// ---------------------------------------------------------------------------
__global__ __launch_bounds__(128) void k_rope_pack(const float* __restrict__ qf,
                                                   const float* __restrict__ kf,
                                                   const float* __restrict__ vf,
                                                   const float* __restrict__ bq,
                                                   const float* __restrict__ bk,
                                                   const float* __restrict__ bv,
                                                   const int* __restrict__ pos_ids,
                                                   __bf16* __restrict__ qb,
                                                   __bf16* __restrict__ kbb,
                                                   __bf16* __restrict__ vtb) {
  const int s = blockIdx.x, hh = blockIdx.y, b = blockIdx.z, d = threadIdx.x;
  const int t = b * Ssz + s;
  const float pos = (float)pos_ids[b * Ssz + s];
  const int i = d & 63;
  const float ang = pos * __powf(1.0e6f, -(float)(2 * i) / 128.f);
  const float c = __cosf(ang), sn = __sinf(ang);
  if (hh < NHn) {
    const float* src = qf + (size_t)t * QN + hh * HDn;
    const float* bb = bq + hh * HDn;
    float x = src[d] + bb[d];
    float xp = src[d ^ 64] + bb[d ^ 64];
    float rh = (d < 64) ? -xp : xp;
    qb[(((size_t)b * NHn + hh) * Ssz + s) * HDn + d] = (__bf16)(x * c + rh * sn);
  } else if (hh < NHn + NKVn) {
    const int kv = hh - NHn;
    const float* src = kf + (size_t)t * KVN + kv * HDn;
    const float* bb = bk + kv * HDn;
    float x = src[d] + bb[d];
    float xp = src[d ^ 64] + bb[d ^ 64];
    float rh = (d < 64) ? -xp : xp;
    kbb[(((size_t)b * NKVn + kv) * Ssz + s) * HDn + d] = (__bf16)(x * c + rh * sn);
  } else {
    const int kv = hh - NHn - NKVn;
    float x = vf[(size_t)t * KVN + kv * HDn + d] + bv[kv * HDn + d];
    vtb[(((size_t)b * NKVn + kv) * HDn + d) * Ssz + s] = (__bf16)x;
  }
}

// ---------------------------------------------------------------------------
// Causal flash attention (GQA). grid: (S/128, NH, B), block 256 (8 waves).
// ---------------------------------------------------------------------------
__global__ __launch_bounds__(256) void k_attention(const __bf16* __restrict__ qb,
                                                   const __bf16* __restrict__ kbb,
                                                   const __bf16* __restrict__ vtb,
                                                   __bf16* __restrict__ attnb) {
  __shared__ __align__(16) __bf16 psmem[8 * 16 * 128];
  const int qt = blockIdx.x, h = blockIdx.y, b = blockIdx.z;
  const int tid = threadIdx.x, wid = tid >> 5, lane = tid & 31;
  const int half = lane >> 4, ln = lane & 15;
  const int kvh = h >> 2;  // NH/NKV = 4
  const __bf16* Q  = qb  + (((size_t)b * NHn  + h)   * Ssz) * HDn;
  const __bf16* Kp = kbb + (((size_t)b * NKVn + kvh) * Ssz) * HDn;
  const __bf16* Vt = vtb + (((size_t)b * NKVn + kvh) * HDn) * Ssz;
  const int q0 = qt * 128 + wid * 16;

  v16bf qa[4];
#pragma unroll
  for (int kk = 0; kk < 4; ++kk) qa[kk] = ld_frag(Q + (size_t)(q0 + ln) * HDn, kk * 32, half);

  float mrow[8], lrow[8];
#pragma unroll
  for (int r = 0; r < 8; ++r) { mrow[r] = -3.0e38f; lrow[r] = 0.f; }
  v8f oacc[8];
#pragma unroll
  for (int n = 0; n < 8; ++n) oacc[n] = zero8();

  __bf16* pt = psmem + wid * (16 * 128);

  for (int kt = 0; kt <= qt; ++kt) {
    const int kb0 = kt * 128;
    v8f sacc[8];
#pragma unroll
    for (int j = 0; j < 8; ++j) sacc[j] = zero8();
#pragma unroll
    for (int j = 0; j < 8; ++j) {
      const __bf16* Brow = Kp + (size_t)(kb0 + j * 16 + ln) * HDn;
#pragma unroll
      for (int kk = 0; kk < 4; ++kk) {
        v16bf bf = ld_frag(Brow, kk * 32, half);
        sacc[j] = wmma_bf16(qa[kk], bf, sacc[j]);
      }
    }
    float pv[8][8];
#pragma unroll
    for (int j = 0; j < 8; ++j)
#pragma unroll
      for (int r = 0; r < 8; ++r) {
        int col = kb0 + j * 16 + ln;
        int row = q0 + half * 8 + r;
        float sv = sacc[j][r] * ATT_SCALE;
        pv[j][r] = (col <= row) ? sv : -3.0e38f;
      }
    float mnew[8], alpha[8];
#pragma unroll
    for (int r = 0; r < 8; ++r) {
      float mv = mrow[r];
#pragma unroll
      for (int j = 0; j < 8; ++j) mv = fmaxf(mv, pv[j][r]);
#pragma unroll
      for (int off = 8; off; off >>= 1) mv = fmaxf(mv, __shfl_xor(mv, off, 16));
      mnew[r] = mv;
      alpha[r] = __expf(mrow[r] - mv);
      mrow[r] = mv;
    }
#pragma unroll
    for (int r = 0; r < 8; ++r) {
      float rs = 0.f;
#pragma unroll
      for (int j = 0; j < 8; ++j) {
        float e = __expf(pv[j][r] - mnew[r]);
        pv[j][r] = e;
        rs += e;
      }
#pragma unroll
      for (int off = 8; off; off >>= 1) rs += __shfl_xor(rs, off, 16);
      lrow[r] = lrow[r] * alpha[r] + rs;
    }
#pragma unroll
    for (int n = 0; n < 8; ++n)
#pragma unroll
      for (int r = 0; r < 8; ++r) oacc[n][r] *= alpha[r];
#pragma unroll
    for (int j = 0; j < 8; ++j)
#pragma unroll
      for (int r = 0; r < 8; ++r)
        pt[(half * 8 + r) * 128 + j * 16 + ln] = (__bf16)pv[j][r];
    v16bf pa[4];
#pragma unroll
    for (int kk = 0; kk < 4; ++kk) pa[kk] = ld_frag(pt + ln * 128, kk * 32, half);
#pragma unroll
    for (int n = 0; n < 8; ++n) {
      const __bf16* Brow = Vt + (size_t)(n * 16 + ln) * Ssz;
#pragma unroll
      for (int kk = 0; kk < 4; ++kk) {
        v16bf bf = ld_frag(Brow, kb0 + kk * 32, half);
        oacc[n] = wmma_bf16(pa[kk], bf, oacc[n]);
      }
    }
  }
#pragma unroll
  for (int n = 0; n < 8; ++n)
#pragma unroll
    for (int r = 0; r < 8; ++r) {
      int row = q0 + half * 8 + r;
      float o = oacc[n][r] / lrow[r];
      attnb[((size_t)(b * Ssz + row)) * QN + h * HDn + n * 16 + ln] = (__bf16)o;
    }
}

// ---------------------------------------------------------------------------
// h = hidden + attn_out + bo ; written straight into d_out.
// ---------------------------------------------------------------------------
__global__ __launch_bounds__(256) void k_resid_add(const float* __restrict__ hidden,
                                                   const float* __restrict__ attno,
                                                   const float* __restrict__ bo,
                                                   float* __restrict__ out) {
  size_t i = (size_t)blockIdx.x * 256 + threadIdx.x;
  int col = (int)(i & (Hsz - 1));
  out[i] = hidden[i] + attno[i] + bo[col];
}

// ---------------------------------------------------------------------------
// Gating: logits = x2 @ gate_w ; top-2 + softmax ; count tokens per expert.
// ---------------------------------------------------------------------------
__global__ __launch_bounds__(256) void k_gate_topk(const __bf16* __restrict__ x2,
                                                   const float* __restrict__ gw,
                                                   int* __restrict__ topidx,
                                                   float* __restrict__ topw,
                                                   int* __restrict__ counts) {
  const int t = blockIdx.x, tid = threadIdx.x;
  float acc[En];
#pragma unroll
  for (int e = 0; e < En; ++e) acc[e] = 0.f;
  for (int hh = tid; hh < Hsz; hh += 256) {
    float xv = (float)x2[(size_t)t * Hsz + hh];
    const float* g = gw + (size_t)hh * En;
#pragma unroll
    for (int e = 0; e < En; ++e) acc[e] += xv * g[e];
  }
#pragma unroll
  for (int e = 0; e < En; ++e)
#pragma unroll
    for (int off = 16; off; off >>= 1) acc[e] += __shfl_xor(acc[e], off, 32);
  __shared__ float red[8][En];
  const int wid = tid >> 5, lane = tid & 31;
  if (lane == 0)
#pragma unroll
    for (int e = 0; e < En; ++e) red[wid][e] = acc[e];
  __syncthreads();
  if (tid == 0) {
    float lg[En];
#pragma unroll
    for (int e = 0; e < En; ++e) {
      float s = 0.f;
#pragma unroll
      for (int w = 0; w < 8; ++w) s += red[w][e];
      lg[e] = s;
    }
    int e0 = 0;
#pragma unroll
    for (int e = 1; e < En; ++e) if (lg[e] > lg[e0]) e0 = e;
    int e1 = (e0 == 0) ? 1 : 0;
#pragma unroll
    for (int e = 0; e < En; ++e) if (e != e0 && lg[e] > lg[e1]) e1 = e;
    float x1 = __expf(lg[e1] - lg[e0]);
    float inv = 1.f / (1.f + x1);
    topidx[t * 2] = e0; topidx[t * 2 + 1] = e1;
    topw[t * 2] = inv;  topw[t * 2 + 1] = x1 * inv;
    atomicAdd(&counts[e0], 1);
    atomicAdd(&counts[e1], 1);
  }
}

__global__ void k_init_counts(int* counts, int* cursor) {
  if (threadIdx.x < En) { counts[threadIdx.x] = 0; cursor[threadIdx.x] = 0; }
}

__global__ void k_prefix_offsets(const int* counts, int* offsets) {
  if (threadIdx.x == 0) {
    int s = 0;
    for (int e = 0; e < En; ++e) { offsets[e] = s; s += counts[e]; }
  }
}

__global__ __launch_bounds__(256) void k_scatter_assign(const int* __restrict__ topidx,
                                                        const float* __restrict__ topw,
                                                        const int* __restrict__ offsets,
                                                        int* __restrict__ cursor,
                                                        int* __restrict__ assign_tok,
                                                        float* __restrict__ assign_w) {
  int t = blockIdx.x * 256 + threadIdx.x;
  if (t >= Tn) return;
#pragma unroll
  for (int s = 0; s < TOPKn; ++s) {
    int e = topidx[t * 2 + s];
    int p = atomicAdd(&cursor[e], 1);
    int a = offsets[e] + p;
    assign_tok[a] = t;
    assign_w[a] = topw[t * 2 + s];
  }
}

// ---------------------------------------------------------------------------
// MoE stage1: act = silu(Xg @ w1[e]) * (Xg @ w3[e])  (gathered rows)
// Double-buffered LDS staging (global_load_b128 + ds_store_b128 path).
// grid: (F/128, T/128, E) ; blocks past count[e] early-exit.
// ---------------------------------------------------------------------------
__global__ __launch_bounds__(256) void k_moe_stage1(const __bf16* __restrict__ x2,
                                                    const __bf16* __restrict__ w1t,
                                                    const __bf16* __restrict__ w3t,
                                                    const int* __restrict__ assign_tok,
                                                    const int* __restrict__ offsets,
                                                    const int* __restrict__ counts,
                                                    __bf16* __restrict__ act) {
  __shared__ __align__(16) __bf16 sA[2][128 * 32];
  __shared__ __align__(16) __bf16 sB1[2][128 * 32];
  __shared__ __align__(16) __bf16 sB3[2][128 * 32];
  const int e = blockIdx.z;
  const int cnt = counts[e];
  const int mt = blockIdx.y;
  if (mt * 128 >= cnt) return;
  const int off = offsets[e];
  const int tid = threadIdx.x, wid = tid >> 5, lane = tid & 31;
  const int half = lane >> 4, ln = lane & 15;
  const int wm = wid & 3, wn = wid >> 2;
  const int n0 = blockIdx.x * 128;

  const int crow = tid >> 2;
  const int cch  = (tid & 3) * 8;
  int rg0 = mt * 128 + crow, rg1 = rg0 + 64;
  int tok0 = assign_tok[off + (rg0 < cnt ? rg0 : cnt - 1)];
  int tok1 = assign_tok[off + (rg1 < cnt ? rg1 : cnt - 1)];
  const __bf16* gA0 = x2 + (size_t)tok0 * Hsz + cch;
  const __bf16* gA1 = x2 + (size_t)tok1 * Hsz + cch;
  const __bf16* w1e = w1t + (size_t)e * Fsz * Hsz;
  const __bf16* w3e = w3t + (size_t)e * Fsz * Hsz;
  const __bf16* gB10 = w1e + (size_t)(n0 + crow) * Hsz + cch;
  const __bf16* gB11 = w1e + (size_t)(n0 + 64 + crow) * Hsz + cch;
  const __bf16* gB30 = w3e + (size_t)(n0 + crow) * Hsz + cch;
  const __bf16* gB31 = w3e + (size_t)(n0 + 64 + crow) * Hsz + cch;

  v8f acc1[2][4], acc3[2][4];
#pragma unroll
  for (int i = 0; i < 2; ++i)
#pragma unroll
    for (int j = 0; j < 4; ++j) { acc1[i][j] = zero8(); acc3[i][j] = zero8(); }

  cp16(gA0, &sA[0][crow * 32 + cch]);
  cp16(gA1, &sA[0][(64 + crow) * 32 + cch]);
  cp16(gB10, &sB1[0][crow * 32 + cch]);
  cp16(gB11, &sB1[0][(64 + crow) * 32 + cch]);
  cp16(gB30, &sB3[0][crow * 32 + cch]);
  cp16(gB31, &sB3[0][(64 + crow) * 32 + cch]);
  __syncthreads();

  int buf = 0;
  for (int k = 0; k < Hsz; k += 32) {
    if (k + 32 < Hsz) {
      const int nb = buf ^ 1, kn = k + 32;
      cp16(gA0 + kn, &sA[nb][crow * 32 + cch]);
      cp16(gA1 + kn, &sA[nb][(64 + crow) * 32 + cch]);
      cp16(gB10 + kn, &sB1[nb][crow * 32 + cch]);
      cp16(gB11 + kn, &sB1[nb][(64 + crow) * 32 + cch]);
      cp16(gB30 + kn, &sB3[nb][crow * 32 + cch]);
      cp16(gB31 + kn, &sB3[nb][(64 + crow) * 32 + cch]);
    }
    const __bf16* pa0 = &sA[buf][(wm * 32 + ln) * 32];
    v16bf a0 = ld_frag(pa0, 0, half);
    v16bf a1 = ld_frag(pa0 + 16 * 32, 0, half);
#pragma unroll
    for (int j = 0; j < 4; ++j) {
      v16bf b1 = ld_frag(&sB1[buf][(wn * 64 + j * 16 + ln) * 32], 0, half);
      acc1[0][j] = wmma_bf16(a0, b1, acc1[0][j]);
      acc1[1][j] = wmma_bf16(a1, b1, acc1[1][j]);
      v16bf b3 = ld_frag(&sB3[buf][(wn * 64 + j * 16 + ln) * 32], 0, half);
      acc3[0][j] = wmma_bf16(a0, b3, acc3[0][j]);
      acc3[1][j] = wmma_bf16(a1, b3, acc3[1][j]);
    }
    __syncthreads();
    buf ^= 1;
  }
#pragma unroll
  for (int i = 0; i < 2; ++i)
#pragma unroll
    for (int j = 0; j < 4; ++j)
#pragma unroll
      for (int r = 0; r < 8; ++r) {
        int row = mt * 128 + wm * 32 + i * 16 + half * 8 + r;
        if (row < cnt) {
          float h1 = acc1[i][j][r], h3 = acc3[i][j][r];
          float sg = h1 / (1.f + __expf(-h1));
          act[(size_t)(off + row) * Fsz + n0 + wn * 64 + j * 16 + ln] =
              (__bf16)(sg * h3);
        }
      }
}

// ---------------------------------------------------------------------------
// MoE stage2: out[token] += routing_w * (act @ w2[e])  (atomic scatter-add)
// grid: (H/128, T/128, E)
// ---------------------------------------------------------------------------
__global__ __launch_bounds__(256) void k_moe_stage2(const __bf16* __restrict__ act,
                                                    const __bf16* __restrict__ w2t,
                                                    const int* __restrict__ assign_tok,
                                                    const float* __restrict__ assign_w,
                                                    const int* __restrict__ offsets,
                                                    const int* __restrict__ counts,
                                                    float* __restrict__ out) {
  __shared__ __align__(16) __bf16 sA[2][128 * 32];
  __shared__ __align__(16) __bf16 sB[2][128 * 32];
  const int e = blockIdx.z;
  const int cnt = counts[e];
  const int mt = blockIdx.y;
  if (mt * 128 >= cnt) return;
  const int off = offsets[e];
  const int tid = threadIdx.x, wid = tid >> 5, lane = tid & 31;
  const int half = lane >> 4, ln = lane & 15;
  const int wm = wid & 3, wn = wid >> 2;
  const int n0 = blockIdx.x * 128;

  const int crow = tid >> 2;
  const int cch  = (tid & 3) * 8;
  int rg0 = mt * 128 + crow, rg1 = rg0 + 64;
  const __bf16* gA0 = act + (size_t)(off + (rg0 < cnt ? rg0 : cnt - 1)) * Fsz + cch;
  const __bf16* gA1 = act + (size_t)(off + (rg1 < cnt ? rg1 : cnt - 1)) * Fsz + cch;
  const __bf16* w2e = w2t + (size_t)e * Hsz * Fsz;
  const __bf16* gB0 = w2e + (size_t)(n0 + crow) * Fsz + cch;
  const __bf16* gB1 = w2e + (size_t)(n0 + 64 + crow) * Fsz + cch;

  v8f acc[2][4];
#pragma unroll
  for (int i = 0; i < 2; ++i)
#pragma unroll
    for (int j = 0; j < 4; ++j) acc[i][j] = zero8();

  cp16(gA0, &sA[0][crow * 32 + cch]);
  cp16(gA1, &sA[0][(64 + crow) * 32 + cch]);
  cp16(gB0, &sB[0][crow * 32 + cch]);
  cp16(gB1, &sB[0][(64 + crow) * 32 + cch]);
  __syncthreads();

  int buf = 0;
  for (int k = 0; k < Fsz; k += 32) {
    if (k + 32 < Fsz) {
      const int nb = buf ^ 1, kn = k + 32;
      cp16(gA0 + kn, &sA[nb][crow * 32 + cch]);
      cp16(gA1 + kn, &sA[nb][(64 + crow) * 32 + cch]);
      cp16(gB0 + kn, &sB[nb][crow * 32 + cch]);
      cp16(gB1 + kn, &sB[nb][(64 + crow) * 32 + cch]);
    }
    const __bf16* pa0 = &sA[buf][(wm * 32 + ln) * 32];
    v16bf a0 = ld_frag(pa0, 0, half);
    v16bf a1 = ld_frag(pa0 + 16 * 32, 0, half);
#pragma unroll
    for (int j = 0; j < 4; ++j) {
      v16bf bf = ld_frag(&sB[buf][(wn * 64 + j * 16 + ln) * 32], 0, half);
      acc[0][j] = wmma_bf16(a0, bf, acc[0][j]);
      acc[1][j] = wmma_bf16(a1, bf, acc[1][j]);
    }
    __syncthreads();
    buf ^= 1;
  }
#pragma unroll
  for (int i = 0; i < 2; ++i)
#pragma unroll
    for (int j = 0; j < 4; ++j)
#pragma unroll
      for (int r = 0; r < 8; ++r) {
        int row = mt * 128 + wm * 32 + i * 16 + half * 8 + r;
        if (row < cnt) {
          int tok = assign_tok[off + row];
          float w = assign_w[off + row];
          atomicAdd(&out[(size_t)tok * Hsz + n0 + wn * 64 + j * 16 + ln],
                    acc[i][j][r] * w);
        }
      }
}

// ---------------------------------------------------------------------------
// Launch
// ---------------------------------------------------------------------------
extern "C" void kernel_launch(void* const* d_in, const int* in_sizes, int n_in,
                              void* d_out, int out_size, void* d_ws, size_t ws_size,
                              hipStream_t stream) {
  const float* hidden = (const float*)d_in[0];
  const int*   pos    = (const int*)d_in[2];
  const float* ln1w = (const float*)d_in[3];
  const float* ln1b = (const float*)d_in[4];
  const float* ln2w = (const float*)d_in[5];
  const float* ln2b = (const float*)d_in[6];
  const float* wq = (const float*)d_in[7];
  const float* bq = (const float*)d_in[8];
  const float* wk = (const float*)d_in[9];
  const float* bk = (const float*)d_in[10];
  const float* wv = (const float*)d_in[11];
  const float* bv = (const float*)d_in[12];
  const float* wo = (const float*)d_in[13];
  const float* bo = (const float*)d_in[14];
  const float* gw = (const float*)d_in[15];
  const float* w1 = (const float*)d_in[16];
  const float* w2 = (const float*)d_in[17];
  const float* w3 = (const float*)d_in[18];
  float* out = (float*)d_out;

  char* p = (char*)d_ws;
  auto carve = [&](size_t elems, size_t esz) -> void* {
    void* r = (void*)p;
    size_t bytes = (elems * esz + 255) & ~(size_t)255;
    p += bytes;
    return r;
  };
  __bf16* wq_t = (__bf16*)carve((size_t)QN * Hsz, 2);
  __bf16* wk_t = (__bf16*)carve((size_t)KVN * Hsz, 2);
  __bf16* wv_t = (__bf16*)carve((size_t)KVN * Hsz, 2);
  __bf16* wo_t = (__bf16*)carve((size_t)Hsz * QN, 2);
  __bf16* w1_t = (__bf16*)carve((size_t)En * Fsz * Hsz, 2);
  __bf16* w3_t = (__bf16*)carve((size_t)En * Fsz * Hsz, 2);
  __bf16* w2_t = (__bf16*)carve((size_t)En * Hsz * Fsz, 2);
  __bf16* x1   = (__bf16*)carve((size_t)Tn * Hsz, 2);
  float*  qf   = (float*)carve((size_t)Tn * QN, 4);
  float*  kf   = (float*)carve((size_t)Tn * KVN, 4);
  float*  vf   = (float*)carve((size_t)Tn * KVN, 4);
  __bf16* qb   = (__bf16*)carve((size_t)Bsz * NHn * Ssz * HDn, 2);
  __bf16* kbb  = (__bf16*)carve((size_t)Bsz * NKVn * Ssz * HDn, 2);
  __bf16* vtb  = (__bf16*)carve((size_t)Bsz * NKVn * HDn * Ssz, 2);
  __bf16* attnb = (__bf16*)carve((size_t)Tn * QN, 2);
  float*  attno = (float*)carve((size_t)Tn * Hsz, 4);
  __bf16* x2   = (__bf16*)carve((size_t)Tn * Hsz, 2);
  int*    topidx = (int*)carve((size_t)Tn * 2, 4);
  float*  topw   = (float*)carve((size_t)Tn * 2, 4);
  int*    counts  = (int*)carve(En, 4);
  int*    offsets = (int*)carve(En, 4);
  int*    cursor  = (int*)carve(En, 4);
  int*    assign_tok = (int*)carve(NAn, 4);
  float*  assign_w   = (float*)carve(NAn, 4);
  __bf16* act = (__bf16*)carve((size_t)NAn * Fsz, 2);

  const dim3 tb(32, 8);
  // 1) weight convert+transpose to bf16 [N,K]
  k_transpose_cvt<<<dim3(QN / 32, Hsz / 32, 1), tb, 0, stream>>>(wq, wq_t, Hsz, QN);
  k_transpose_cvt<<<dim3(KVN / 32, Hsz / 32, 1), tb, 0, stream>>>(wk, wk_t, Hsz, KVN);
  k_transpose_cvt<<<dim3(KVN / 32, Hsz / 32, 1), tb, 0, stream>>>(wv, wv_t, Hsz, KVN);
  k_transpose_cvt<<<dim3(Hsz / 32, QN / 32, 1), tb, 0, stream>>>(wo, wo_t, QN, Hsz);
  k_transpose_cvt<<<dim3(Fsz / 32, Hsz / 32, En), tb, 0, stream>>>(w1, w1_t, Hsz, Fsz);
  k_transpose_cvt<<<dim3(Fsz / 32, Hsz / 32, En), tb, 0, stream>>>(w3, w3_t, Hsz, Fsz);
  k_transpose_cvt<<<dim3(Hsz / 32, Fsz / 32, En), tb, 0, stream>>>(w2, w2_t, Fsz, Hsz);
  // 2) LN1 + QKV projections
  k_layernorm_bf16<<<Tn, 256, 0, stream>>>(hidden, ln1w, ln1b, x1);
  k_gemm_bf16_f32<true><<<dim3(QN / 128, Tn / 128), 256, 0, stream>>>(x1, wq_t, bq, qf, QN, Hsz);
  k_gemm_bf16_f32<true><<<dim3(KVN / 128, Tn / 128), 256, 0, stream>>>(x1, wk_t, bk, kf, KVN, Hsz);
  k_gemm_bf16_f32<true><<<dim3(KVN / 128, Tn / 128), 256, 0, stream>>>(x1, wv_t, bv, vf, KVN, Hsz);
  // 3) RoPE + repack
  k_rope_pack<<<dim3(Ssz, NHn + 2 * NKVn, Bsz), 128, 0, stream>>>(qf, kf, vf, bq, bk, bv,
                                                                  pos, qb, kbb, vtb);
  // 4) attention
  k_attention<<<dim3(Ssz / 128, NHn, Bsz), 256, 0, stream>>>(qb, kbb, vtb, attnb);
  // 5) out-proj + residual + LN2
  k_gemm_bf16_f32<false><<<dim3(Hsz / 128, Tn / 128), 256, 0, stream>>>(attnb, wo_t, nullptr,
                                                                        attno, Hsz, QN);
  k_resid_add<<<(Tn * Hsz) / 256, 256, 0, stream>>>(hidden, attno, bo, out);
  k_layernorm_bf16<<<Tn, 256, 0, stream>>>(out, ln2w, ln2b, x2);
  // 6) routing
  k_init_counts<<<1, 64, 0, stream>>>(counts, cursor);
  k_gate_topk<<<Tn, 256, 0, stream>>>(x2, gw, topidx, topw, counts);
  k_prefix_offsets<<<1, 32, 0, stream>>>(counts, offsets);
  k_scatter_assign<<<Tn / 256, 256, 0, stream>>>(topidx, topw, offsets, cursor,
                                                 assign_tok, assign_w);
  // 7) expert GEMMs (sparse top-2)
  k_moe_stage1<<<dim3(Fsz / 128, Tn / 128, En), 256, 0, stream>>>(x2, w1_t, w3_t, assign_tok,
                                                                  offsets, counts, act);
  k_moe_stage2<<<dim3(Hsz / 128, Tn / 128, En), 256, 0, stream>>>(act, w2_t, assign_tok,
                                                                  assign_w, offsets, counts,
                                                                  out);
}
